// RGCNLayer_7559142441674
// MI455X (gfx1250) — matrix operand order repaired and checked by reading, not compiled, not run on hardware
//
#include <hip/hip_runtime.h>

typedef __attribute__((ext_vector_type(16))) __bf16        v16bf;
typedef __attribute__((ext_vector_type(8)))  float         v8f;
typedef __attribute__((ext_vector_type(4)))  float         f32x4;
typedef __attribute__((ext_vector_type(4)))  unsigned int  u32x4;
typedef __attribute__((ext_vector_type(2)))  unsigned int  u32x2;

#define DIM     128
#define WPAD    136      // padded LDS row (bf16 elems): 272B -> conflict-free strided reads
#define WAVES   8
#define THREADS 256

__device__ __forceinline__ unsigned short f2bf(float f) {
  unsigned int u = __float_as_uint(f);
  u += 0x7FFFu + ((u >> 16) & 1u);      // round-to-nearest-even
  return (unsigned short)(u >> 16);
}

// packed f32x2 -> bf16x2 (RNE), single VALU op on CDNA5
__device__ __forceinline__ unsigned int pk_bf16(float lo, float hi) {
  unsigned int r;
  asm("v_cvt_pk_bf16_f32 %0, %1, %2" : "=v"(r) : "v"(lo), "v"(hi));
  return r;
}

__global__ void zero_kernel(float* __restrict__ out, int n) {
  int i = blockIdx.x * blockDim.x + threadIdx.x;
  int stride = gridDim.x * blockDim.x;
  for (; i < n; i += stride) out[i] = 0.f;
}

// Wt[r][o][i] = bf16( sum_b A[r,b] * V[b,i,o] )  -- stored transposed (out-major)
__global__ void compute_W(const float* __restrict__ A, const float* __restrict__ V,
                          unsigned short* __restrict__ Wt, int Nr, int B) {
  int idx = blockIdx.x * blockDim.x + threadIdx.x;
  int total = Nr * DIM * DIM;
  if (idx >= total) return;
  int r   = idx / (DIM * DIM);
  int rem = idx - r * DIM * DIM;
  int o   = rem >> 7;
  int i   = rem & (DIM - 1);
  float acc = 0.f;
  for (int b = 0; b < B; ++b)
    acc += A[r * B + b] * V[((size_t)b * DIM + i) * DIM + o];
  Wt[idx] = f2bf(acc);
}

// D^T formulation:  D[o_tile(16) x edges(16)] = Wt_tile(A, 16x32) x Htile(B, 32x16)
//   A-operand: W rows (out-major, double-buffered from LDS) -> M = output channel
//   B-operand: gathered H rows (regs all tile)              -> N = edge index == lane&15
__global__ __launch_bounds__(THREADS) void rgcn_edge_wmma(
    const float* __restrict__ H,
    const int*   __restrict__ edge_row,
    const int*   __restrict__ edge_col,
    const float* __restrict__ edge_val,
    const unsigned short* __restrict__ Wt,
    float* __restrict__ out,
    int E, int chunksPerRel)
{
  __shared__ __align__(16) unsigned short Wlds[DIM * WPAD];         // 34 KB: W_r^T, bf16
  __shared__ __align__(16) unsigned short Hlds[WAVES * 16 * WPAD];  // 34 KB: per-wave H tiles

  const int r     = blockIdx.x / chunksPerRel;
  const int chunk = blockIdx.x % chunksPerRel;

  // ---- stage Wt[r] into LDS (16B chunks, padded rows) ----
  {
    const unsigned short* wsrc = Wt + (size_t)r * DIM * DIM;
    for (int idx = threadIdx.x; idx < (DIM * DIM) / 8; idx += THREADS) {
      int o = idx >> 4;       // 16 x 16B chunks per 128-elem row
      int c = idx & 15;
      u32x4 v = ((const u32x4*)(wsrc + o * DIM))[c];
      *((u32x4*)&Wlds[o * WPAD + c * 8]) = v;
    }
  }
  __syncthreads();

  const int wave = threadIdx.x >> 5;
  const int lane = threadIdx.x & 31;
  const int m    = lane & 15;     // A-row within half (out chan) / B-col (edge idx)
  const int hi   = lane >> 4;     // lane half
  unsigned short* hst = &Hlds[wave * 16 * WPAD];

  const size_t rE = (size_t)r * (size_t)E;
  const int tilesPerRel   = (E + 15) >> 4;
  const int tilesPerChunk = (tilesPerRel + chunksPerRel - 1) / chunksPerRel;
  const int tStart = chunk * tilesPerChunk;
  const int tEnd   = (tStart + tilesPerChunk < tilesPerRel) ? (tStart + tilesPerChunk)
                                                            : tilesPerRel;

  union FR { u32x4 u[2]; v16bf v; };

  for (int t = tStart + wave; t < tEnd; t += WAVES) {
    // ---- edge metadata: lanes 0..15 each own one edge of the tile ----
    // tail edges: row=0, val=0 -> branchless (atomicAdd of 0.0f is a no-op)
    int e = t * 16 + m;
    int col = 0, drow = 0; float val = 0.f;
    if (hi == 0 && e < E) {
      col  = edge_col[rE + e];
      drow = edge_row[rE + e];
      val  = edge_val[rE + e];
    }

    // ---- gather 16 H rows (f32, L2-resident), convert to bf16, stage ----
    #pragma unroll
    for (int rr = 0; rr < 16; ++rr) {
      int c = __shfl(col, rr, 32);
      f32x4 h = *(const f32x4*)(H + (size_t)c * DIM + lane * 4);
      u32x2 pk;
      pk.x = pk_bf16(h.x, h.y);
      pk.y = pk_bf16(h.z, h.w);
      *((u32x2*)&hst[rr * WPAD + lane * 4]) = pk;
    }
    __builtin_amdgcn_wave_barrier();
    asm volatile("" ::: "memory");

    // ---- broadcast this lane's edge (col = lane&15) meta to both halves ----
    float vme = __shfl(val,  m, 32);
    int   dre = __shfl(drow, m, 32);
    float* obase = out + (size_t)dre * DIM + hi * 8;   // per-lane output base

    // ---- B fragments: H tile as 32x16 (K x edges), kept in regs all tile ----
    FR bfr[4];
    #pragma unroll
    for (int ks = 0; ks < 4; ++ks) {
      const u32x4* bp = (const u32x4*)&hst[m * WPAD + ks * 32 + hi * 8];
      bfr[ks].u[0] = bp[0];      // K = ks*32 + hi*8 .. +7, then ks*32 + 16 + hi*8 .. +7 at edge column m
      bfr[ks].u[1] = bp[2];
    }

    // ---- 8 output-channel tiles; W fragments double-buffered from LDS ----
    FR afA[4], afB[4];
    #pragma unroll
    for (int ks = 0; ks < 4; ++ks) {           // preload nt=0
      const u32x4* ap = (const u32x4*)&Wlds[m * WPAD + ks * 32 + hi * 8];
      afA[ks].u[0] = ap[0];
      afA[ks].u[1] = ap[2];
    }

    #pragma unroll
    for (int nt = 0; nt < 8; ++nt) {
      FR* cur = (nt & 1) ? afB : afA;
      FR* nxt = (nt & 1) ? afA : afB;
      if (nt < 7) {                            // prefetch W rows for nt+1
        #pragma unroll
        for (int ks = 0; ks < 4; ++ks) {
          const u32x4* ap =
              (const u32x4*)&Wlds[((nt + 1) * 16 + m) * WPAD + ks * 32 + hi * 8];
          nxt[ks].u[0] = ap[0];   // K = ks*32 + hi*8      .. +7   (VGPR 0-3)
          nxt[ks].u[1] = ap[2];   // K = ks*32 + 16 + hi*8 .. +7   (VGPR 4-7)
        }
      }
      v8f acc = {0.f, 0.f, 0.f, 0.f, 0.f, 0.f, 0.f, 0.f};
      #pragma unroll
      for (int ks = 0; ks < 4; ++ks) {
        acc = __builtin_amdgcn_wmma_f32_16x16x32_bf16(
            false, cur[ks].v, false, bfr[ks].v, (short)0, acc, false, false);
      }
      // ---- scale by edge_val, scatter-add: immediate-offset atomics ----
      #pragma unroll
      for (int j = 0; j < 8; ++j) {
        // output channel = nt*16 + hi*8 + j; hi*8 folded into obase
        atomicAdd(obase + nt * 16 + j, acc[j] * vme);
      }
    }
    __builtin_amdgcn_wave_barrier();
    asm volatile("" ::: "memory");
  }
}

extern "C" void kernel_launch(void* const* d_in, const int* in_sizes, int n_in,
                              void* d_out, int out_size, void* d_ws, size_t ws_size,
                              hipStream_t stream) {
  const float* H    = (const float*)d_in[0];
  const float* A    = (const float*)d_in[1];
  const float* V    = (const float*)d_in[2];
  const int*   erow = (const int*)d_in[3];
  const int*   ecol = (const int*)d_in[4];
  const float* ev   = (const float*)d_in[5];
  float* out = (float*)d_out;
  unsigned short* Wt = (unsigned short*)d_ws;   // needs Nr*128*128*2 = 256 KB

  int B  = in_sizes[2] / (DIM * DIM);           // 4 bases
  int Nr = (B > 0) ? in_sizes[1] / B : 1;       // 8 relations
  int E  = (Nr > 0) ? in_sizes[3] / Nr : 0;     // 200000 edges/relation

  zero_kernel<<<2048, 256, 0, stream>>>(out, out_size);

  int wtot = Nr * DIM * DIM;
  compute_W<<<(wtot + 255) / 256, 256, 0, stream>>>(A, V, Wt, Nr, B);

  int chunks = 100;                             // 800 blocks x 8 waves
  rgcn_edge_wmma<<<Nr * chunks, THREADS, 0, stream>>>(H, erow, ecol, ev, Wt, out,
                                                      E, chunks);
}
